// EGCL_90280212562508
// MI455X (gfx1250) — matrix operand order repaired and checked by reading, not compiled, not run on hardware
//
#include <hip/hip_runtime.h>

#define BB  2
#define NN  384
#define HH  128
#define NCH 8      // j-chunks
#define JCH 48     // columns per chunk (3 x 16)

typedef __attribute__((ext_vector_type(16))) _Float16 v16h;
typedef __attribute__((ext_vector_type(8)))  _Float16 v8h;
typedef __attribute__((ext_vector_type(8)))  float    v8f;

// Fast SiLU: x * v_rcp_f32(1 + exp(-x)) -- 4 VALU ops, no IEEE div expansion.
__device__ __forceinline__ float silu_f(float x) {
  const float e = __expf(-x);                       // v_exp_f32
  return x * __builtin_amdgcn_rcpf(1.0f + e);       // v_add + v_rcp + v_mul
}

__device__ __forceinline__ v16h cat8(v8h a, v8h b) {
  return __builtin_shufflevector(a, b, 0,1,2,3,4,5,6,7,8,9,10,11,12,13,14,15);
}

// ---------------------------------------------------------------------------
// Kernel 1: per-node precompute  A = feat@Wm1[0:128] + 0.5*bm1,
//                                Bn = feat@Wm1[128:256] + 0.5*bm1
// ---------------------------------------------------------------------------
__global__ __launch_bounds__(128) void egcl_prep(
    const float* __restrict__ feat, const float* __restrict__ Wm1,
    const float* __restrict__ bm1, float* __restrict__ Ag, float* __restrict__ Bg)
{
  const int row = blockIdx.x;
  const int h = threadIdx.x;
  __shared__ float f[HH];
  f[h] = feat[(long)row*HH + h];
  __syncthreads();
  float a = 0.5f * bm1[h];
  float b = a;
  #pragma unroll 4
  for (int d = 0; d < HH; ++d) {
    const float fv = f[d];
    a = fmaf(fv, Wm1[d*HH + h], a);
    b = fmaf(fv, Wm1[(HH + d)*HH + h], b);
  }
  Ag[(long)row*HH + h] = a;
  Bg[(long)row*HH + h] = b;
}

// ---------------------------------------------------------------------------
// Kernel 2: fused pairwise message/coord kernel (WMMA f16, message never
// touches HBM). Block = 16 i-rows x 48 j-cols, 8 waves, wave owns 2 i-rows.
// ---------------------------------------------------------------------------
__global__ __launch_bounds__(256, 1) void egcl_pair(
    const float* __restrict__ radial, const float* __restrict__ disp,
    const float* __restrict__ adj,    const float* __restrict__ mask2d,
    const float* __restrict__ Ag,     const float* __restrict__ Bg,
    const float* __restrict__ wr,     const float* __restrict__ bm2,
    const float* __restrict__ Wm2,    const float* __restrict__ Wc1,
    const float* __restrict__ bc1,    const float* __restrict__ Wc2,
    float* __restrict__ agg_out,      float* __restrict__ coord_out)
{
  extern __shared__ char smem[];
  _Float16* WmT  = (_Float16*)(smem);            // [128][136] f16 : Wm2 transposed
  _Float16* WcT  = (_Float16*)(smem + 34816);    // [128][136] f16 : Wc1 transposed
  _Float16* msgS = (_Float16*)(smem + 69632);    // 8 waves * [16][136] f16
  float* Ablk = (float*)(smem + 104448);         // [16][132] f32
  float* Bblk = (float*)(smem + 112896);         // [48][132] f32
  float* wrS  = (float*)(smem + 138240);
  float* bm2S = (float*)(smem + 138752);
  float* bc1S = (float*)(smem + 139264);
  float* wc2S = (float*)(smem + 139776);
  float* w2dS = (float*)(smem + 140288);         // 8 waves * 16

  const int b    = blockIdx.z;
  const int i0   = blockIdx.x * 16;
  const int j0   = blockIdx.y * JCH;
  const int tid  = threadIdx.x;
  const int lane = tid & 31;
  const int wave = tid >> 5;
  const int m    = lane & 15;
  const int half = lane >> 4;

  // Stage weights (transposed, f16) + activations (f32) into LDS.
  for (int e = tid; e < HH*HH; e += 256) {
    const int k = e >> 7, n = e & 127;
    WmT[n*136 + k] = (_Float16)Wm2[e];
    WcT[n*136 + k] = (_Float16)Wc1[e];
  }
  for (int e = tid; e < 16*HH; e += 256) {
    const int r = e >> 7, c = e & 127;
    Ablk[r*132 + c] = Ag[((long)(b*NN + i0 + r))*HH + c];
  }
  for (int e = tid; e < JCH*HH; e += 256) {
    const int r = e >> 7, c = e & 127;
    Bblk[r*132 + c] = Bg[((long)(b*NN + j0 + r))*HH + c];
  }
  if (tid < HH) { wrS[tid]=wr[tid]; bm2S[tid]=bm2[tid]; bc1S[tid]=bc1[tid]; wc2S[tid]=Wc2[tid]; }
  __syncthreads();

  // Hoist loop-invariant per-lane bias/weight scalars out of the hot loop.
  float bmv[8], bcv[8], wcv[8];
  #pragma unroll
  for (int nt = 0; nt < 8; ++nt) {
    bmv[nt] = bm2S[nt*16 + m];
    bcv[nt] = bc1S[nt*16 + m];
    wcv[nt] = wc2S[nt*16 + m];
  }

  _Float16* msgW = msgS + wave * (16*136);
  float aggp[2][8];
  #pragma unroll
  for (int a1 = 0; a1 < 2; ++a1)
    #pragma unroll
    for (int a2 = 0; a2 < 8; ++a2) aggp[a1][a2] = 0.f;
  float csum[2][3] = {{0.f,0.f,0.f},{0.f,0.f,0.f}};

  const v8f vzero = {0.f,0.f,0.f,0.f,0.f,0.f,0.f,0.f};

  for (int jt = 0; jt < 3; ++jt) {
    const int jb = jt * 16;
    const int jg = j0 + jb;
    for (int tt = 0; tt < 2; ++tt) {
      const int t = wave*2 + tt;          // i-row within tile (wave-owned)
      const int i = i0 + t;
      const int j = jg + m;               // this lane's j column (M row of tile)
      const long pidx = ((long)(b*NN + i))*NN + j;
      const float r_l = radial[pidx];
      if (lane < 16) w2dS[wave*16 + m] = mask2d[pidx] * adj[pidx];
      __syncthreads();

      // agg weights for this iteration: 8 consecutive floats -> two b128 loads
      const float4 w2a = *(const float4*)(w2dS + wave*16 + 8*half);
      const float4 w2b = *(const float4*)(w2dS + wave*16 + 8*half + 4);
      const float w2v[8] = { w2a.x, w2a.y, w2a.z, w2a.w, w2b.x, w2b.y, w2b.z, w2b.w };

      const float* arow = Ablk + t*132;
      const float* brow = Bblk + (jb + m)*132;

      // ---- GEMM1: pre2 = h1 @ Wm2, h1 built on the fly in A-layout ----
      v8f acc[8];
      #pragma unroll
      for (int nt = 0; nt < 8; ++nt) acc[nt] = vzero;
      #pragma unroll
      for (int ks = 0; ks < 4; ++ks) {
        const int g = ks*32 + half*8;     // ISA A-layout: two groups of 8 K
        v16h af;
        #pragma unroll
        for (int grp = 0; grp < 2; ++grp) {
          const int k0 = g + grp*16;
          const float4 a0 = *(const float4*)(arow + k0);
          const float4 a1 = *(const float4*)(arow + k0 + 4);
          const float4 b0 = *(const float4*)(brow + k0);
          const float4 b1 = *(const float4*)(brow + k0 + 4);
          const float4 w0 = *(const float4*)(wrS + k0);
          const float4 w1 = *(const float4*)(wrS + k0 + 4);
          const int o = grp*8;
          af[o+0] = (_Float16)silu_f(fmaf(r_l, w0.x, a0.x + b0.x));
          af[o+1] = (_Float16)silu_f(fmaf(r_l, w0.y, a0.y + b0.y));
          af[o+2] = (_Float16)silu_f(fmaf(r_l, w0.z, a0.z + b0.z));
          af[o+3] = (_Float16)silu_f(fmaf(r_l, w0.w, a0.w + b0.w));
          af[o+4] = (_Float16)silu_f(fmaf(r_l, w1.x, a1.x + b1.x));
          af[o+5] = (_Float16)silu_f(fmaf(r_l, w1.y, a1.y + b1.y));
          af[o+6] = (_Float16)silu_f(fmaf(r_l, w1.z, a1.z + b1.z));
          af[o+7] = (_Float16)silu_f(fmaf(r_l, w1.w, a1.w + b1.w));
        }
        #pragma unroll
        for (int nt = 0; nt < 8; ++nt) {
          const _Float16* wp = WmT + (nt*16 + m)*136 + ks*32 + half*8;
          const v16h bf = cat8(*(const v8h*)wp, *(const v8h*)(wp + 16));
          acc[nt] = __builtin_amdgcn_wmma_f32_16x16x32_f16(
              false, af, false, bf, (short)0, acc[nt], false, false);
        }
      }

      // message = silu(pre2 + bm2): stage to LDS + agg partials (registers)
      #pragma unroll
      for (int nt = 0; nt < 8; ++nt) {
        #pragma unroll
        for (int r = 0; r < 8; ++r) {
          const float mg = silu_f(acc[nt][r] + bmv[nt]); // C-layout: M=r+8*half, N=nt*16+m
          msgW[(r + 8*half)*136 + nt*16 + m] = (_Float16)mg;
          aggp[tt][nt] += mg * w2v[r];
        }
      }
      __syncthreads();

      // ---- GEMM2: cc = message @ Wc1 ----
      v8f cc[8];
      #pragma unroll
      for (int nt = 0; nt < 8; ++nt) cc[nt] = vzero;
      #pragma unroll
      for (int ks = 0; ks < 4; ++ks) {
        const int g = ks*32 + half*8;
        const _Float16* mp = msgW + m*136;
        const v16h af = cat8(*(const v8h*)(mp + g), *(const v8h*)(mp + g + 16));
        #pragma unroll
        for (int nt = 0; nt < 8; ++nt) {
          const _Float16* wp = WcT + (nt*16 + m)*136 + ks*32 + half*8;
          const v16h bf = cat8(*(const v8h*)wp, *(const v8h*)(wp + 16));
          cc[nt] = __builtin_amdgcn_wmma_f32_16x16x32_f16(
              false, af, false, bf, (short)0, cc[nt], false, false);
        }
      }

      // u[jj] = sum_h silu(cc + bc1)*Wc2 : lane-partial then wave32 reduction
      float up[8];
      #pragma unroll
      for (int r = 0; r < 8; ++r) up[r] = 0.f;
      #pragma unroll
      for (int nt = 0; nt < 8; ++nt) {
        #pragma unroll
        for (int r = 0; r < 8; ++r) up[r] += silu_f(cc[nt][r] + bcv[nt]) * wcv[nt];
      }
      #pragma unroll
      for (int r = 0; r < 8; ++r) {
        up[r] += __shfl_xor(up[r], 1, 32);
        up[r] += __shfl_xor(up[r], 2, 32);
        up[r] += __shfl_xor(up[r], 4, 32);
        up[r] += __shfl_xor(up[r], 8, 32);
      }
      float u2[8];
      #pragma unroll
      for (int r = 0; r < 8; ++r) u2[r] = __shfl(up[r], 16, 32);  // half1 -> lane0

      if (lane == 0) {  // deterministic coord accumulation, lane0 owns row i
        #pragma unroll
        for (int q = 0; q < 16; ++q) {
          const float uv = (q < 8) ? up[q] : u2[q-8];
          const int j2 = jg + q;
          if (j2 != i) {
            const long qidx = ((long)(b*NN + i))*NN + j2;
            const float m2v = mask2d[qidx];
            const float* dp = disp + qidx*3;
            #pragma unroll
            for (int x = 0; x < 3; ++x) {
              float v = uv * dp[x];
              v = fminf(100.0f, fmaxf(-100.0f, v));
              csum[tt][x] += v * m2v * m2v;   // mask2d applied twice, as in ref
            }
          }
        }
      }
      __syncthreads();
    }
  }

  // Write per-chunk partials (slabs -> deterministic reduce in finalize).
  #pragma unroll
  for (int tt = 0; tt < 2; ++tt) {
    const int i = i0 + wave*2 + tt;
    const long base = ((long)(b*NCH + blockIdx.y))*NN + i;
    #pragma unroll
    for (int nt = 0; nt < 8; ++nt) {
      const float v = aggp[tt][nt] + __shfl_xor(aggp[tt][nt], 16, 32);
      if (lane < 16) agg_out[base*HH + nt*16 + lane] = v;
    }
    if (lane == 0) {
      #pragma unroll
      for (int x = 0; x < 3; ++x) coord_out[base*3 + x] = csum[tt][x];
    }
  }
}

// ---------------------------------------------------------------------------
// Kernel 3: reduce chunk partials + per-node feat MLP + coord scaling.
// ---------------------------------------------------------------------------
__global__ __launch_bounds__(128) void egcl_final(
    const float* __restrict__ feat, const float* __restrict__ mask,
    const float* __restrict__ mask2d,
    const float* __restrict__ aggp, const float* __restrict__ coordp,
    const float* __restrict__ Wf1, const float* __restrict__ bf1,
    const float* __restrict__ Wf2, const float* __restrict__ bf2,
    float* __restrict__ outf, float* __restrict__ outc)
{
  const int row = blockIdx.x;          // b*N + i
  const int b = row / NN, i = row - b*NN;
  const int h = threadIdx.x;
  __shared__ float nf[2*HH];
  __shared__ float hid[HH];

  float ag = 0.f;
  for (int c = 0; c < NCH; ++c) ag += aggp[(((long)(b*NCH + c))*NN + i)*HH + h];
  nf[h]      = feat[(long)row*HH + h];
  nf[HH + h] = ag;
  __syncthreads();

  float a1 = bf1[h];
  #pragma unroll 4
  for (int k = 0; k < 2*HH; ++k) a1 = fmaf(nf[k], Wf1[k*HH + h], a1);
  hid[h] = silu_f(a1);
  __syncthreads();

  float o = bf2[h];
  #pragma unroll 4
  for (int k = 0; k < HH; ++k) o = fmaf(hid[k], Wf2[k*HH + h], o);
  const float mk = mask[row];
  outf[(long)row*HH + h] = o * mk;

  if (h < 3) {
    float cs = 0.f;
    for (int c = 0; c < NCH; ++c) cs += coordp[(((long)(b*NCH + c))*NN + i)*3 + h];
    float C = 0.f;                                  // C = sum_j m2[b,0,j]
    for (int j = 1; j < NN; ++j) C += mask2d[(long)b*NN*NN + j];
    outc[(long)row*3 + h] = C * cs * mk;
  }
}

// ---------------------------------------------------------------------------
extern "C" void kernel_launch(void* const* d_in, const int* in_sizes, int n_in,
                              void* d_out, int out_size, void* d_ws, size_t ws_size,
                              hipStream_t stream) {
  (void)in_sizes; (void)n_in; (void)out_size; (void)ws_size;
  const float* feat   = (const float*)d_in[0];
  /* d_in[1] = coord (unused by reference math) */
  const float* radial = (const float*)d_in[2];
  const float* disp   = (const float*)d_in[3];
  const float* adj    = (const float*)d_in[4];
  const float* mask   = (const float*)d_in[5];
  const float* mask2d = (const float*)d_in[6];
  const float* Wm1 = (const float*)d_in[7];
  const float* bm1 = (const float*)d_in[8];
  const float* Wm2 = (const float*)d_in[9];
  const float* bm2 = (const float*)d_in[10];
  const float* Wc1 = (const float*)d_in[11];
  const float* bc1 = (const float*)d_in[12];
  const float* Wc2 = (const float*)d_in[13];
  const float* Wf1 = (const float*)d_in[14];
  const float* bf1 = (const float*)d_in[15];
  const float* Wf2 = (const float*)d_in[16];
  const float* bf2 = (const float*)d_in[17];

  float* ws     = (float*)d_ws;
  float* Ag     = ws;                                  // B*N*H
  float* Bg     = Ag + (long)BB*NN*HH;                 // B*N*H
  float* aggp   = Bg + (long)BB*NN*HH;                 // B*NCH*N*H
  float* coordp = aggp + (long)BB*NCH*NN*HH;           // B*NCH*N*3

  float* outf = (float*)d_out;
  float* outc = outf + (long)BB*NN*HH;

  egcl_prep<<<dim3(BB*NN), dim3(128), 0, stream>>>(feat, Wm1, bm1, Ag, Bg);
  egcl_pair<<<dim3(NN/16, NCH, BB), dim3(256), 140800, stream>>>(
      radial, disp, adj, mask2d, Ag, Bg, Wm1 + 256*HH, bm2, Wm2, Wc1, bc1, Wc2,
      aggp, coordp);
  egcl_final<<<dim3(BB*NN), dim3(128), 0, stream>>>(
      feat, mask, mask2d, aggp, coordp, Wf1, bf1, Wf2, bf2, outf, outc);
}
